// GINLayer_83167746719884
// MI455X (gfx1250) — matrix operand order repaired and checked by reading, not compiled, not run on hardware
//
#include <hip/hip_runtime.h>
#include <hip/hip_bf16.h>

// ---------------------------------------------------------------------------
// GIN layer for MI455X (gfx1250, wave32):
//   agg = (1+eps)*x + segment_sum(x[col], row)
//   h   = agg @ W1^T + b1 ; BN(train) ; ReLU
//   out = h @ W2^T + b2
// Memory-bound: ~1 GB traffic vs 6.6 GFLOP. GEMMs use v_wmma_f32_16x16x32_f16
// (f16 inputs, f32 accumulate). Scatter uses native global_atomic_add_f32;
// both x and agg fit in the 192 MB L2, so gather/atomics run at L2 speed.
// ---------------------------------------------------------------------------

typedef __attribute__((ext_vector_type(16))) _Float16 v16h;
typedef __attribute__((ext_vector_type(8)))  _Float16 v8h;
typedef __attribute__((ext_vector_type(4)))  _Float16 h4;
typedef __attribute__((ext_vector_type(8)))  float    v8f;

#define FDIM 128
#define LDSW 136   // padded LDS row stride (halves) for the weight tile

// ---------------------------------------------------------------------------
// Kernel 1: agg = (1+eps)*x ; zero BN stat accumulators
// ---------------------------------------------------------------------------
__global__ __launch_bounds__(256)
void gin_init_kernel(const float* __restrict__ x, const float* __restrict__ eps,
                     float* __restrict__ agg, float* __restrict__ stats, int total4)
{
    if (blockIdx.x == 0 && threadIdx.x < 256) stats[threadIdx.x] = 0.0f;
    int i = blockIdx.x * blockDim.x + threadIdx.x;
    if (i < total4) {
        float c = 1.0f + eps[0];
        float4 v = ((const float4*)x)[i];
        v.x *= c; v.y *= c; v.z *= c; v.w *= c;
        ((float4*)agg)[i] = v;
    }
}

// ---------------------------------------------------------------------------
// Kernel 2: scatter-add. One wave per edge; lane l moves bytes [16l,16l+16) of
// the 512B source row (fully coalesced gather) and does 4 f32 atomic adds.
// ---------------------------------------------------------------------------
__global__ __launch_bounds__(256)
void gin_scatter_kernel(const float* __restrict__ x, const long long* __restrict__ ei,
                        float* __restrict__ agg, int E)
{
    int gtid = blockIdx.x * blockDim.x + threadIdx.x;
    int e    = gtid >> 5;
    if (e >= E) return;
    int lane = threadIdx.x & 31;
    long long dst = ei[e];                 // row (destination node)
    long long src = ei[(long long)E + e];  // col (source node)
    float4 v = *((const float4*)(x + (size_t)src * FDIM) + lane);
    float* ap = agg + (size_t)dst * FDIM + lane * 4;
    __hip_atomic_fetch_add(ap + 0, v.x, __ATOMIC_RELAXED, __HIP_MEMORY_SCOPE_AGENT);
    __hip_atomic_fetch_add(ap + 1, v.y, __ATOMIC_RELAXED, __HIP_MEMORY_SCOPE_AGENT);
    __hip_atomic_fetch_add(ap + 2, v.z, __ATOMIC_RELAXED, __HIP_MEMORY_SCOPE_AGENT);
    __hip_atomic_fetch_add(ap + 3, v.w, __ATOMIC_RELAXED, __HIP_MEMORY_SCOPE_AGENT);
}

// ---------------------------------------------------------------------------
// Helpers: build f16 A-fragments per the CDNA5 16-bit 16x32 A layout.
// lane: m = lane&15, hi = lane>>4. For K-tile kt, halves j=0..7 hold
// K = kt*32 + hi*8 + j ; halves j=8..15 hold K = kt*32 + 16 + hi*8 + (j-8).
// ---------------------------------------------------------------------------
__device__ inline void cvt8(v16h& a, int base, float4 u0, float4 u1)
{
    a[base+0] = (_Float16)u0.x; a[base+1] = (_Float16)u0.y;
    a[base+2] = (_Float16)u0.z; a[base+3] = (_Float16)u0.w;
    a[base+4] = (_Float16)u1.x; a[base+5] = (_Float16)u1.y;
    a[base+6] = (_Float16)u1.z; a[base+7] = (_Float16)u1.w;
}

__device__ inline void cvt8_bnrelu(v16h& a, int base, float4 u0, float4 u1,
                                   const float* __restrict__ sc,
                                   const float* __restrict__ sh, int k)
{
    float f[8] = {u0.x, u0.y, u0.z, u0.w, u1.x, u1.y, u1.z, u1.w};
#pragma unroll
    for (int j = 0; j < 8; ++j) {
        float t = f[j] * sc[k + j] + sh[k + j];
        a[base + j] = (_Float16)fmaxf(t, 0.0f);
    }
}

// Load weight matrix (row-major f32 [128][128]) into LDS as f16, padded rows.
__device__ inline void load_w_lds(_Float16* __restrict__ w, const float* __restrict__ W)
{
    const int tid = threadIdx.x;
#pragma unroll
    for (int i = 0; i < 16; ++i) {
        int q   = tid + 256 * i;   // float4 index, 4096 total
        int row = q >> 5;          // 32 float4 per row
        int c4  = q & 31;
        float4 wv = ((const float4*)(W + row * FDIM))[c4];
        h4 p;
        p.x = (_Float16)wv.x; p.y = (_Float16)wv.y;
        p.z = (_Float16)wv.z; p.w = (_Float16)wv.w;
        *(h4*)(&w[row * LDSW + c4 * 4]) = p;
    }
}

// B fragment: dense 16-bit 32x16 layout => lane (col = nbase + lane&15) needs
// K halves kt*32 + (lane>>4)*8 + j (j<8) and kt*32 + 16 + (lane>>4)*8 + (j-8) of W row `col`.
__device__ inline v16h load_b_frag(const _Float16* __restrict__ w, int col, int kb)
{
    v8h lo = *(const v8h*)(&w[col * LDSW + kb]);
    v8h hi = *(const v8h*)(&w[col * LDSW + kb + 16]);
    v16h b;
#pragma unroll
    for (int j = 0; j < 8; ++j) { b[j] = lo[j]; b[j + 8] = hi[j]; }
    return b;
}

// ---------------------------------------------------------------------------
// Kernel 3: h = agg @ W1^T + b1 (in-place over agg) + BN partial stats.
// Block = 256 thr = 8 waves; wave w owns rows [blk*128 + w*16, +16).
// ---------------------------------------------------------------------------
__global__ __launch_bounds__(256)
void gin_gemm1_kernel(float* __restrict__ buf, const float* __restrict__ W1,
                      const float* __restrict__ b1, float* __restrict__ gsum,
                      float* __restrict__ gsq, int Nn)
{
    __shared__ _Float16 w[FDIM * LDSW];
    __shared__ float s_sum[FDIM], s_sq[FDIM];

    const int tid = threadIdx.x;
    load_w_lds(w, W1);
    if (tid < FDIM) { s_sum[tid] = 0.0f; s_sq[tid] = 0.0f; }
    __syncthreads();

    const int wave = tid >> 5, lane = tid & 31;
    const int m = lane & 15, hi = lane >> 4;
    const int rowbase = (blockIdx.x * 8 + wave) * 16;
    const bool full = (rowbase + 16 <= Nn);     // wave-uniform: scalar branch

    int arow = rowbase + m; if (arow > Nn - 1) arow = Nn - 1;   // clamp tail
    const float* ap = buf + (size_t)arow * FDIM;

    v16h afrag[4];
#pragma unroll
    for (int kt = 0; kt < 4; ++kt) {
        int k0 = kt * 32 + hi * 8;
        float4 u0 = *(const float4*)(ap + k0);
        float4 u1 = *(const float4*)(ap + k0 + 4);
        float4 u2 = *(const float4*)(ap + k0 + 16);
        float4 u3 = *(const float4*)(ap + k0 + 20);
        v16h a;
        cvt8(a, 0, u0, u1);
        cvt8(a, 8, u2, u3);
        afrag[kt] = a;
    }

#pragma unroll
    for (int nt = 0; nt < 8; ++nt) {
        const int col = nt * 16 + m;
        v8f acc = {0.f, 0.f, 0.f, 0.f, 0.f, 0.f, 0.f, 0.f};
#pragma unroll
        for (int kt = 0; kt < 4; ++kt) {
            v16h b = load_b_frag(w, col, kt * 32 + hi * 8);
            acc = __builtin_amdgcn_wmma_f32_16x16x32_f16(
                false, afrag[kt], false, b, (short)0, acc, false, false);
        }
        float bc = b1[col];
        float s = 0.0f, sq = 0.0f;
        // C layout: lane covers rows rowbase + 8*hi + r, fixed col.
        float* p = buf + (size_t)(rowbase + 8 * hi) * FDIM + col;
        if (full) {
#pragma unroll
            for (int r = 0; r < 8; ++r) {
                float v = acc[r] + bc;
                p[r * FDIM] = v;
                s += v; sq += v * v;
            }
        } else {
#pragma unroll
            for (int r = 0; r < 8; ++r) {
                int row = rowbase + r + 8 * hi;
                float v = acc[r] + bc;
                if (row < Nn) {
                    p[r * FDIM] = v;
                    s += v; sq += v * v;
                }
            }
        }
        __hip_atomic_fetch_add(&s_sum[col], s,  __ATOMIC_RELAXED, __HIP_MEMORY_SCOPE_WORKGROUP);
        __hip_atomic_fetch_add(&s_sq[col],  sq, __ATOMIC_RELAXED, __HIP_MEMORY_SCOPE_WORKGROUP);
    }
    __syncthreads();
    if (tid < FDIM) {
        __hip_atomic_fetch_add(&gsum[tid], s_sum[tid], __ATOMIC_RELAXED, __HIP_MEMORY_SCOPE_AGENT);
        __hip_atomic_fetch_add(&gsq[tid],  s_sq[tid],  __ATOMIC_RELAXED, __HIP_MEMORY_SCOPE_AGENT);
    }
}

// ---------------------------------------------------------------------------
// Kernel 4: fold BN stats into per-feature affine: scale/shift
// ---------------------------------------------------------------------------
__global__ void gin_bnfin_kernel(const float* __restrict__ gsum, const float* __restrict__ gsq,
                                 const float* __restrict__ gamma, const float* __restrict__ beta,
                                 float* __restrict__ scale, float* __restrict__ shift, int Nn)
{
    int i = threadIdx.x;
    if (i < FDIM) {
        float inv_n = 1.0f / (float)Nn;
        float mean  = gsum[i] * inv_n;
        float var   = gsq[i] * inv_n - mean * mean;
        float inv   = rsqrtf(var + 1e-5f);
        float sc    = gamma[i] * inv;
        scale[i] = sc;
        shift[i] = beta[i] - mean * sc;
    }
}

// ---------------------------------------------------------------------------
// Kernel 5: out = relu(h*scale+shift) @ W2^T + b2. BN+ReLU fused into the
// f32->f16 A-fragment conversion.
// ---------------------------------------------------------------------------
__global__ __launch_bounds__(256)
void gin_gemm2_kernel(const float* __restrict__ hbuf, const float* __restrict__ W2,
                      const float* __restrict__ b2, const float* __restrict__ scale,
                      const float* __restrict__ shift, float* __restrict__ out, int Nn)
{
    __shared__ _Float16 w[FDIM * LDSW];
    __shared__ float s_sc[FDIM], s_sh[FDIM];

    const int tid = threadIdx.x;
    load_w_lds(w, W2);
    if (tid < FDIM) { s_sc[tid] = scale[tid]; s_sh[tid] = shift[tid]; }
    __syncthreads();

    const int wave = tid >> 5, lane = tid & 31;
    const int m = lane & 15, hi = lane >> 4;
    const int rowbase = (blockIdx.x * 8 + wave) * 16;
    const bool full = (rowbase + 16 <= Nn);     // wave-uniform: scalar branch

    int arow = rowbase + m; if (arow > Nn - 1) arow = Nn - 1;
    const float* ap = hbuf + (size_t)arow * FDIM;

    v16h afrag[4];
#pragma unroll
    for (int kt = 0; kt < 4; ++kt) {
        int k0 = kt * 32 + hi * 8;
        int k1 = k0 + 16;
        float4 u0 = *(const float4*)(ap + k0);
        float4 u1 = *(const float4*)(ap + k0 + 4);
        float4 u2 = *(const float4*)(ap + k1);
        float4 u3 = *(const float4*)(ap + k1 + 4);
        v16h a;
        cvt8_bnrelu(a, 0, u0, u1, s_sc, s_sh, k0);
        cvt8_bnrelu(a, 8, u2, u3, s_sc, s_sh, k1);
        afrag[kt] = a;
    }

#pragma unroll
    for (int nt = 0; nt < 8; ++nt) {
        const int col = nt * 16 + m;
        v8f acc = {0.f, 0.f, 0.f, 0.f, 0.f, 0.f, 0.f, 0.f};
#pragma unroll
        for (int kt = 0; kt < 4; ++kt) {
            v16h b = load_b_frag(w, col, kt * 32 + hi * 8);
            acc = __builtin_amdgcn_wmma_f32_16x16x32_f16(
                false, afrag[kt], false, b, (short)0, acc, false, false);
        }
        float bc = b2[col];
        float* p = out + (size_t)(rowbase + 8 * hi) * FDIM + col;
        if (full) {
#pragma unroll
            for (int r = 0; r < 8; ++r) {
                p[r * FDIM] = acc[r] + bc;
            }
        } else {
#pragma unroll
            for (int r = 0; r < 8; ++r) {
                int row = rowbase + r + 8 * hi;
                if (row < Nn) p[r * FDIM] = acc[r] + bc;
            }
        }
    }
}

// ---------------------------------------------------------------------------
// Launch. Inputs: x, edge_index(i64), eps, W1, b1, gamma, beta, W2, b2.
// Workspace: buf[N*128] (agg, overwritten in-place with h) | sum[128] |
// sq[128] | scale[128] | shift[128].
// ---------------------------------------------------------------------------
extern "C" void kernel_launch(void* const* d_in, const int* in_sizes, int n_in,
                              void* d_out, int out_size, void* d_ws, size_t ws_size,
                              hipStream_t stream)
{
    const float*     x     = (const float*)d_in[0];
    const long long* ei    = (const long long*)d_in[1];
    const float*     eps   = (const float*)d_in[2];
    const float*     W1    = (const float*)d_in[3];
    const float*     b1    = (const float*)d_in[4];
    const float*     gamma = (const float*)d_in[5];
    const float*     beta  = (const float*)d_in[6];
    const float*     W2    = (const float*)d_in[7];
    const float*     b2    = (const float*)d_in[8];
    float*           out   = (float*)d_out;

    const int Nn = in_sizes[0] / FDIM;   // 100000
    const int E  = in_sizes[1] / 2;      // 1600000

    float* buf   = (float*)d_ws;
    float* gsum  = buf + (size_t)Nn * FDIM;
    float* gsq   = gsum + FDIM;
    float* scale = gsq + FDIM;
    float* shift = scale + FDIM;

    // 1) agg = (1+eps)*x ; zero stats
    int total4 = Nn * (FDIM / 4);
    gin_init_kernel<<<(total4 + 255) / 256, 256, 0, stream>>>(x, eps, buf, gsum, total4);

    // 2) scatter-add over edges (1 wave per edge)
    long long sthreads = (long long)E * 32;
    gin_scatter_kernel<<<(int)((sthreads + 255) / 256), 256, 0, stream>>>(x, ei, buf, E);

    // 3) GEMM1 + BN partial stats (h written in-place over agg)
    int mblocks = (Nn + 127) / 128;
    gin_gemm1_kernel<<<mblocks, 256, 0, stream>>>(buf, W1, b1, gsum, gsq, Nn);

    // 4) fold stats into affine
    gin_bnfin_kernel<<<1, 128, 0, stream>>>(gsum, gsq, gamma, beta, scale, shift, Nn);

    // 5) GEMM2 with fused BN+ReLU on A
    gin_gemm2_kernel<<<mblocks, 256, 0, stream>>>(buf, W2, b2, scale, shift, out, Nn);
}
